// TemporalAxialAttention_34376918237760
// MI455X (gfx1250) — compile-verified, matrix operand order checked by reading, not run on hardware
//
#include <hip/hip_runtime.h>
#include <math.h>

// ---------------- shapes ----------------
#define BATCH   4
#define TT_     20
#define HH_     24
#define WW_     24
#define DMODEL  512
#define IMODEL  512
#define TWIN    10
#define NBLK    2            // TT_/TWIN
#define BHW     2304         // BATCH*HH_*WW_
#define HEADS   8
#define DHEAD   64
#define QKVN    1536         // 3*IMODEL
#define GATEN   2048         // 4*IMODEL

// ---------------- WMMA types ----------------
typedef __attribute__((ext_vector_type(16))) __bf16 v16bf;
typedef __attribute__((ext_vector_type(8)))  float  v8f;

union Frag { v16bf v; unsigned u[8]; };

// round-to-nearest fp32->bf16 pair pack: 2x v_add + 1x v_perm_b32
__device__ __forceinline__ unsigned pack2(float lo, float hi) {
    const unsigned ul = __float_as_uint(lo) + 0x8000u;
    const unsigned uh = __float_as_uint(hi) + 0x8000u;
    return __builtin_amdgcn_perm(uh, ul, 0x07060302u);  // {uh[31:16], ul[31:16]}
}
__device__ __forceinline__ v8f wmma_bf16(const Frag& a, const Frag& b, v8f c) {
    return __builtin_amdgcn_wmma_f32_16x16x32_bf16(
        /*neg_a=*/false, a.v, /*neg_b=*/false, b.v,
        /*c_mod=*/(short)0, c, /*reuse_a=*/false, /*reuse_b=*/false);
}
__device__ __forceinline__ float sigm(float x) { return 1.f / (1.f + expf(-x)); }

// B fragment (bf16 [N,K] row-major): p = dword ptr at row start + kc*16
__device__ __forceinline__ void load_b_frag(Frag& f, const unsigned* p, int lhalf) {
    const uint4 lo = *(const uint4*)(p + 4 * lhalf);
    const uint4 hi = *(const uint4*)(p + 8 + 4 * lhalf);
    f.u[0] = lo.x; f.u[1] = lo.y; f.u[2] = lo.z; f.u[3] = lo.w;
    f.u[4] = hi.x; f.u[5] = hi.y; f.u[6] = hi.z; f.u[7] = hi.w;
}
// A fragment from fp32 row: p = float ptr at row start + kc*32
__device__ __forceinline__ void load_a_frag_f32(Frag& f, const float* p, int lhalf) {
    const float4 a0 = *(const float4*)(p + 8 * lhalf);
    const float4 a1 = *(const float4*)(p + 8 * lhalf + 4);
    const float4 b0 = *(const float4*)(p + 16 + 8 * lhalf);
    const float4 b1 = *(const float4*)(p + 16 + 8 * lhalf + 4);
    f.u[0] = pack2(a0.x, a0.y); f.u[1] = pack2(a0.z, a0.w);
    f.u[2] = pack2(a1.x, a1.y); f.u[3] = pack2(a1.z, a1.w);
    f.u[4] = pack2(b0.x, b0.y); f.u[5] = pack2(b0.z, b0.w);
    f.u[6] = pack2(b1.x, b1.y); f.u[7] = pack2(b1.z, b1.w);
}

// ---------------- weight fp32 -> bf16 (vectorized) ----------------
__global__ __launch_bounds__(256) void convert_bf16_kernel(
        const float4* __restrict__ src, uint2* __restrict__ dst, int n4) {
    const int i = blockIdx.x * 256 + threadIdx.x;
    if (i < n4) {
        const float4 v = src[i];
        dst[i] = make_uint2(pack2(v.x, v.y), pack2(v.z, v.w));
    }
}

__global__ __launch_bounds__(256) void zero_kernel(uint4* __restrict__ p, int n16) {
    const int i = blockIdx.x * 256 + threadIdx.x;
    if (i < n16) p[i] = make_uint4(0u, 0u, 0u, 0u);
}

// ---------------- fused QKV GEMM + RoPE + causal attention ----------------
// one block per (n, bhw); 8 waves. GEMM: each wave owns 3 groups of 4 N-tiles
// (16x64 C per group) -> A fragment loaded once per K-chunk feeds 4 WMMAs.
// Padding rows 10..15 use clamped in-bounds addresses; their C rows are never
// stored (WMMA C rows depend only on the matching A row).
__global__ __launch_bounds__(256) void qkv_attn_kernel(
        const float* __restrict__ x, const unsigned* __restrict__ wq,
        float* __restrict__ o) {
    __shared__ float qkv_s[TWIN * QKVN];   // 61440 B
    __shared__ float sc_s[HEADS * 112];    //  3584 B  (total 65024 <= 64KB)

    const int blk  = blockIdx.x;          // 0..NBLK*BHW-1
    const int n    = blk / BHW;
    const int bhw  = blk % BHW;
    const int b    = bhw / (HH_ * WW_);
    const int hw   = bhw % (HH_ * WW_);
    const int hh   = hw / WW_;
    const int ww   = hw % WW_;
    const int tid  = threadIdx.x;
    const int wave = tid >> 5;
    const int lane = tid & 31;
    const int lrow = lane & 15;
    const int lhalf = lane >> 4;

    // A row lrow = timestep; clamp so padding rows stay in-bounds
    int t = n * TWIN + lrow;
    if (t > TT_ - 1) t = TT_ - 1;
    const float* xrow = x + (((b * TT_ + t) * HH_ + hh) * WW_ + ww) * DMODEL;

    for (int g = 0; g < 3; ++g) {                    // 3 groups of 4 tiles
        const int ntbase = (g * 8 + wave) * 4;
        const unsigned* wrow0 = wq + ((ntbase + 0) * 16 + lrow) * (DMODEL / 2);
        const unsigned* wrow1 = wq + ((ntbase + 1) * 16 + lrow) * (DMODEL / 2);
        const unsigned* wrow2 = wq + ((ntbase + 2) * 16 + lrow) * (DMODEL / 2);
        const unsigned* wrow3 = wq + ((ntbase + 3) * 16 + lrow) * (DMODEL / 2);
        v8f c0 = {}, c1 = {}, c2 = {}, c3 = {};
#pragma unroll 2
        for (int kc = 0; kc < DMODEL / 32; ++kc) {
            __builtin_prefetch(wrow0 + (kc + 4) * 16, 0, 1);
            Frag a, bf;
            load_a_frag_f32(a, xrow + kc * 32, lhalf);
            load_b_frag(bf, wrow0 + kc * 16, lhalf);  c0 = wmma_bf16(a, bf, c0);
            load_b_frag(bf, wrow1 + kc * 16, lhalf);  c1 = wmma_bf16(a, bf, c1);
            load_b_frag(bf, wrow2 + kc * 16, lhalf);  c2 = wmma_bf16(a, bf, c2);
            load_b_frag(bf, wrow3 + kc * 16, lhalf);  c3 = wmma_bf16(a, bf, c3);
        }
#pragma unroll
        for (int r = 0; r < 8; ++r) {
            const int row = r + 8 * lhalf;
            if (row < TWIN) {
                float* q = qkv_s + row * QKVN + ntbase * 16 + lrow;
                q[0]  = c0[r];
                q[16] = c1[r];
                q[32] = c2[r];
                q[48] = c3[r];
            }
        }
    }
    __syncthreads();

    // ---- attention: wave = head ----
    const int head = wave;
    {   // RoPE on q and k; lane owns pair j = lane
        const int j = lane;                       // 0..31 pairs of DHEAD
        const float inv = powf(10000.f, -((float)(2 * j)) / (float)DHEAD);
        for (int tt = 0; tt < TWIN; ++tt) {
            const float ang = (float)tt * inv;
            const float cs = cosf(ang), sn = sinf(ang);
#pragma unroll
            for (int sel = 0; sel < 2; ++sel) {   // q then k
                const int col = sel * IMODEL + head * DHEAD + 2 * j;
                const float x0 = qkv_s[tt * QKVN + col];
                const float x1 = qkv_s[tt * QKVN + col + 1];
                qkv_s[tt * QKVN + col]     = x0 * cs - x1 * sn;
                qkv_s[tt * QKVN + col + 1] = x1 * cs + x0 * sn;
            }
        }
    }
    __syncthreads();

    for (int e = lane; e < TWIN * TWIN; e += 32) {
        const int qt = e / TWIN, kt = e % TWIN;
        float s = -3.0e38f;
        if (kt <= qt) {
            float acc = 0.f;
            for (int d = 0; d < DHEAD; ++d)
                acc += qkv_s[qt * QKVN + head * DHEAD + d] *
                       qkv_s[kt * QKVN + IMODEL + head * DHEAD + d];
            s = acc * 0.125f;                     // DHEAD^-0.5
        }
        sc_s[head * 112 + e] = s;
    }
    __syncthreads();

    if (lane < TWIN) {                            // softmax row qt=lane
        const int qt = lane;
        float mx = -3.0e38f;
        for (int kt = 0; kt <= qt; ++kt)
            mx = fmaxf(mx, sc_s[head * 112 + qt * TWIN + kt]);
        float ev[TWIN], sum = 0.f;
        for (int kt = 0; kt <= qt; ++kt) {
            ev[kt] = expf(sc_s[head * 112 + qt * TWIN + kt] - mx);
            sum += ev[kt];
        }
        const float rs = 1.f / sum;
        for (int kt = 0; kt < TWIN; ++kt)
            sc_s[head * 112 + qt * TWIN + kt] = (kt <= qt) ? ev[kt] * rs : 0.f;
    }
    __syncthreads();

    for (int idx = lane; idx < TWIN * DHEAD; idx += 32) {
        const int qt = idx / DHEAD, d = idx % DHEAD;
        float acc = 0.f;
        for (int kt = 0; kt <= qt; ++kt)
            acc += sc_s[head * 112 + qt * TWIN + kt] *
                   qkv_s[kt * QKVN + 2 * IMODEL + head * DHEAD + d];
        o[((n * BHW + bhw) * TWIN + qt) * IMODEL + head * DHEAD + d] = acc;
    }
}

// ---------------- LSTM: gates = [xt|h] @ [w_ih|w_hh]^T  (one timestep) -------
// wave = one (mt, 4-N-tile group): 144 * 32 units / 8 waves = 576 blocks
__global__ __launch_bounds__(256) void lstm_gates_kernel(
        const float* __restrict__ o, const unsigned* __restrict__ hbf,
        const unsigned* __restrict__ wih, const unsigned* __restrict__ whh,
        float* __restrict__ gates, int n, int tt) {
    const int unit  = blockIdx.x * 8 + (threadIdx.x >> 5);   // 0..4607
    const int lane  = threadIdx.x & 31;
    const int lrow  = lane & 15;
    const int lhalf = lane >> 4;
    const int mt = unit >> 5;          // /32
    const int ng = unit & 31;          // N group of 4 tiles
    const int m  = mt * 16 + lrow;
    const int wrbase = ng * 64 + lrow; // (ng*4)*16 + lrow
    const float*    orow = o   + ((n * BHW + m) * TWIN + tt) * IMODEL;
    const unsigned* hrow = hbf + m * (IMODEL / 2);
    const unsigned* wi0 = wih + (wrbase)      * (IMODEL / 2);
    const unsigned* wi1 = wih + (wrbase + 16) * (IMODEL / 2);
    const unsigned* wi2 = wih + (wrbase + 32) * (IMODEL / 2);
    const unsigned* wi3 = wih + (wrbase + 48) * (IMODEL / 2);
    const unsigned* wh0 = whh + (wrbase)      * (IMODEL / 2);
    const unsigned* wh1 = whh + (wrbase + 16) * (IMODEL / 2);
    const unsigned* wh2 = whh + (wrbase + 32) * (IMODEL / 2);
    const unsigned* wh3 = whh + (wrbase + 48) * (IMODEL / 2);

    v8f c0 = {}, c1 = {}, c2 = {}, c3 = {};
#pragma unroll 2
    for (int kc = 0; kc < IMODEL / 32; ++kc) {       // x-part
        __builtin_prefetch(wi0 + (kc + 4) * 16, 0, 1);
        Frag a, bf;
        load_a_frag_f32(a, orow + kc * 32, lhalf);
        load_b_frag(bf, wi0 + kc * 16, lhalf);  c0 = wmma_bf16(a, bf, c0);
        load_b_frag(bf, wi1 + kc * 16, lhalf);  c1 = wmma_bf16(a, bf, c1);
        load_b_frag(bf, wi2 + kc * 16, lhalf);  c2 = wmma_bf16(a, bf, c2);
        load_b_frag(bf, wi3 + kc * 16, lhalf);  c3 = wmma_bf16(a, bf, c3);
    }
#pragma unroll 2
    for (int kc = 0; kc < IMODEL / 32; ++kc) {       // h-part
        __builtin_prefetch(wh0 + (kc + 4) * 16, 0, 1);
        Frag a, bf;
        load_b_frag(a, hrow + kc * 16, lhalf);
        load_b_frag(bf, wh0 + kc * 16, lhalf);  c0 = wmma_bf16(a, bf, c0);
        load_b_frag(bf, wh1 + kc * 16, lhalf);  c1 = wmma_bf16(a, bf, c1);
        load_b_frag(bf, wh2 + kc * 16, lhalf);  c2 = wmma_bf16(a, bf, c2);
        load_b_frag(bf, wh3 + kc * 16, lhalf);  c3 = wmma_bf16(a, bf, c3);
    }
#pragma unroll
    for (int r = 0; r < 8; ++r) {
        float* gp = gates + (mt * 16 + r + 8 * lhalf) * GATEN + ng * 64 + lrow;
        gp[0]  = c0[r];
        gp[16] = c1[r];
        gp[32] = c2[r];
        gp[48] = c3[r];
    }
}

// ---------------- LSTM pointwise: c,h update; y = o + h (in place over o) ----
__global__ __launch_bounds__(256) void lstm_point_kernel(
        const float* __restrict__ gates, const float* __restrict__ b_ih,
        const float* __restrict__ b_hh, float* __restrict__ y,
        unsigned* __restrict__ hbf, float* __restrict__ c,
        int n, int tt) {
    const int i   = blockIdx.x * 256 + threadIdx.x;  // < BHW*IMODEL/4
    const int m   = i >> 7;                          // /128
    const int j4  = (i & 127) * 4;
    const float* g = gates + m * GATEN + j4;
    const float4 I = *(const float4*)(g);
    const float4 F = *(const float4*)(g + IMODEL);
    const float4 G = *(const float4*)(g + 2 * IMODEL);
    const float4 O = *(const float4*)(g + 3 * IMODEL);
    const float4 bi0 = *(const float4*)(b_ih + j4);
    const float4 bi1 = *(const float4*)(b_ih + IMODEL + j4);
    const float4 bi2 = *(const float4*)(b_ih + 2 * IMODEL + j4);
    const float4 bi3 = *(const float4*)(b_ih + 3 * IMODEL + j4);
    const float4 bh0 = *(const float4*)(b_hh + j4);
    const float4 bh1 = *(const float4*)(b_hh + IMODEL + j4);
    const float4 bh2 = *(const float4*)(b_hh + 2 * IMODEL + j4);
    const float4 bh3 = *(const float4*)(b_hh + 3 * IMODEL + j4);
    float4 co = *(float4*)(c + m * IMODEL + j4);

    float hv[4], cv[4];
    const float ig[4] = { I.x + bi0.x + bh0.x, I.y + bi0.y + bh0.y,
                          I.z + bi0.z + bh0.z, I.w + bi0.w + bh0.w };
    const float fg[4] = { F.x + bi1.x + bh1.x, F.y + bi1.y + bh1.y,
                          F.z + bi1.z + bh1.z, F.w + bi1.w + bh1.w };
    const float gg[4] = { G.x + bi2.x + bh2.x, G.y + bi2.y + bh2.y,
                          G.z + bi2.z + bh2.z, G.w + bi2.w + bh2.w };
    const float og[4] = { O.x + bi3.x + bh3.x, O.y + bi3.y + bh3.y,
                          O.z + bi3.z + bh3.z, O.w + bi3.w + bh3.w };
    const float cold[4] = { co.x, co.y, co.z, co.w };
#pragma unroll
    for (int q = 0; q < 4; ++q) {
        cv[q] = sigm(fg[q]) * cold[q] + sigm(ig[q]) * tanhf(gg[q]);
        hv[q] = sigm(og[q]) * tanhf(cv[q]);
    }
    *(float4*)(c + m * IMODEL + j4) = make_float4(cv[0], cv[1], cv[2], cv[3]);
    *(uint2*)(hbf + (m * IMODEL + j4) / 2) =
        make_uint2(pack2(hv[0], hv[1]), pack2(hv[2], hv[3]));
    float* yp = y + ((n * BHW + m) * TWIN + tt) * IMODEL + j4;
    const float4 yo = *(const float4*)(yp);
    *(float4*)(yp) = make_float4(yo.x + hv[0], yo.y + hv[1],
                                 yo.z + hv[2], yo.w + hv[3]);
}

// ---------------- output projection: out = y' @ w_out^T + b_out -------------
// wave = one (mt, 4-N-tile group): 2880 * 8 units / 8 waves = 2880 blocks
__global__ __launch_bounds__(256) void out_proj_kernel(
        const float* __restrict__ y, const unsigned* __restrict__ wout,
        const float* __restrict__ bout, float* __restrict__ out) {
    const int unit  = blockIdx.x * 8 + (threadIdx.x >> 5);   // 0..23039
    const int lane  = threadIdx.x & 31;
    const int lrow  = lane & 15;
    const int lhalf = lane >> 4;
    const int mt = unit >> 3;        // /8
    const int ng = unit & 7;         // N group of 4 tiles
    const int m  = mt * 16 + lrow;
    // out row m = ((b*T + t)*H + hh)*W + ww  ->  y[n, b*576+hw, tt, :]
    const int b  = m / (TT_ * HH_ * WW_);
    const int r1 = m % (TT_ * HH_ * WW_);
    const int t  = r1 / (HH_ * WW_);
    const int hw = r1 % (HH_ * WW_);
    const int n  = t / TWIN, tt = t % TWIN;
    const float* yrow = y + ((n * BHW + b * HH_ * WW_ + hw) * TWIN + tt) * IMODEL;
    const int wrbase = ng * 64 + lrow;
    const unsigned* w0 = wout + (wrbase)      * (IMODEL / 2);
    const unsigned* w1 = wout + (wrbase + 16) * (IMODEL / 2);
    const unsigned* w2 = wout + (wrbase + 32) * (IMODEL / 2);
    const unsigned* w3 = wout + (wrbase + 48) * (IMODEL / 2);

    v8f c0 = {}, c1 = {}, c2 = {}, c3 = {};
#pragma unroll 2
    for (int kc = 0; kc < IMODEL / 32; ++kc) {
        __builtin_prefetch(w0 + (kc + 4) * 16, 0, 1);
        Frag a, bf;
        load_a_frag_f32(a, yrow + kc * 32, lhalf);
        load_b_frag(bf, w0 + kc * 16, lhalf);  c0 = wmma_bf16(a, bf, c0);
        load_b_frag(bf, w1 + kc * 16, lhalf);  c1 = wmma_bf16(a, bf, c1);
        load_b_frag(bf, w2 + kc * 16, lhalf);  c2 = wmma_bf16(a, bf, c2);
        load_b_frag(bf, w3 + kc * 16, lhalf);  c3 = wmma_bf16(a, bf, c3);
    }
    const int nc0 = ng * 64 + lrow;
    const float b0 = bout[nc0], b1 = bout[nc0 + 16],
                b2 = bout[nc0 + 32], b3 = bout[nc0 + 48];
#pragma unroll
    for (int r = 0; r < 8; ++r) {
        float* op = out + (mt * 16 + r + 8 * lhalf) * DMODEL + nc0;
        op[0]  = c0[r] + b0;
        op[16] = c1[r] + b1;
        op[32] = c2[r] + b2;
        op[48] = c3[r] + b3;
    }
}

// ---------------- host-side orchestration ----------------
extern "C" void kernel_launch(void* const* d_in, const int* in_sizes, int n_in,
                              void* d_out, int out_size, void* d_ws, size_t ws_size,
                              hipStream_t stream) {
    const float* x     = (const float*)d_in[0];
    const float* w_qkv = (const float*)d_in[1];
    const float* w_out = (const float*)d_in[2];
    const float* b_out = (const float*)d_in[3];
    const float* w_ih  = (const float*)d_in[4];
    const float* w_hh  = (const float*)d_in[5];
    const float* b_ih  = (const float*)d_in[6];
    const float* b_hh  = (const float*)d_in[7];
    float* out = (float*)d_out;

    char* base = (char*)d_ws;
    size_t off = 0;
    auto carve = [&](size_t bytes) -> void* {
        void* p = base + off;
        off = (off + bytes + 255) & ~(size_t)255;
        return p;
    };
    unsigned short* wq_bf   = (unsigned short*)carve((size_t)QKVN * DMODEL * 2);
    unsigned short* wih_bf  = (unsigned short*)carve((size_t)GATEN * IMODEL * 2);
    unsigned short* whh_bf  = (unsigned short*)carve((size_t)GATEN * IMODEL * 2);
    unsigned short* wout_bf = (unsigned short*)carve((size_t)DMODEL * IMODEL * 2);
    float* obuf  = (float*)carve((size_t)NBLK * BHW * TWIN * IMODEL * 4);  // o, then y in place
    float* gates = (float*)carve((size_t)BHW * GATEN * 4);
    float* cbuf  = (float*)carve((size_t)BHW * IMODEL * 4);
    unsigned* hbf = (unsigned*)carve((size_t)BHW * IMODEL * 2);

    // 1) weights -> bf16 (L2-resident thereafter)
    convert_bf16_kernel<<<(QKVN * DMODEL / 4 + 255) / 256, 256, 0, stream>>>(
        (const float4*)w_qkv, (uint2*)wq_bf, QKVN * DMODEL / 4);
    convert_bf16_kernel<<<(GATEN * IMODEL / 4 + 255) / 256, 256, 0, stream>>>(
        (const float4*)w_ih, (uint2*)wih_bf, GATEN * IMODEL / 4);
    convert_bf16_kernel<<<(GATEN * IMODEL / 4 + 255) / 256, 256, 0, stream>>>(
        (const float4*)w_hh, (uint2*)whh_bf, GATEN * IMODEL / 4);
    convert_bf16_kernel<<<(DMODEL * IMODEL / 4 + 255) / 256, 256, 0, stream>>>(
        (const float4*)w_out, (uint2*)wout_bf, DMODEL * IMODEL / 4);

    // 2) zero c and h_bf (contiguous carve, both multiples of 16B)
    const int z16 = (BHW * IMODEL * 4 + BHW * IMODEL * 2) / 16;
    zero_kernel<<<(z16 + 255) / 256, 256, 0, stream>>>((uint4*)cbuf, z16);

    // 3) fused QKV GEMM + RoPE + causal attention
    qkv_attn_kernel<<<NBLK * BHW, 256, 0, stream>>>(
        x, (const unsigned*)wq_bf, obuf);

    // 4) LSTM: 20 sequential steps (carry across the two blocks)
    for (int s = 0; s < NBLK * TWIN; ++s) {
        const int n = s / TWIN, tt = s % TWIN;
        lstm_gates_kernel<<<(144 * 32) / 8, 256, 0, stream>>>(
            obuf, (const unsigned*)hbf, (const unsigned*)wih_bf,
            (const unsigned*)whh_bf, gates, n, tt);
        lstm_point_kernel<<<(BHW * IMODEL / 4) / 256, 256, 0, stream>>>(
            gates, b_ih, b_hh, obuf, hbf, cbuf, n, tt);
    }

    // 5) output projection
    out_proj_kernel<<<(2880 * 8) / 8, 256, 0, stream>>>(
        obuf, (const unsigned*)wout_bf, b_out, out);

    (void)in_sizes; (void)n_in; (void)out_size; (void)ws_size;
}
